// CrossScaleAttention_29308856828526
// MI455X (gfx1250) — matrix operand order repaired and checked by reading, not compiled, not run on hardware
//
#include <hip/hip_runtime.h>
#include <math.h>

// ---------------- problem constants ----------------
#define BB    4
#define NPIX  4096      // 64*64
#define EDIM  64
#define ODIM  256
#define HDCH  256
#define LDCH  512

typedef __attribute__((ext_vector_type(16))) __bf16 v16bf;
typedef __attribute__((ext_vector_type(8)))  float  v8f;

static __device__ __forceinline__ v8f wmma_bf16(v16bf a, v16bf b, v8f c) {
  // D = A(16x32) * B(32x16) + C, f32 accumulate
  return __builtin_amdgcn_wmma_f32_16x16x32_bf16(false, a, false, b, (short)0, c, false, false);
}

union FragU { v16bf v; uint4 u[2]; };
union Pack8 { uint4 q; __bf16 e[8]; };

// A-fragment: per-lane row, elements K = {2*kb..2*kb+15} contiguous  (kb = 8*(lane>=16))
static __device__ __forceinline__ v16bf load_a_frag(const __bf16* rowk0, int kb) {
  FragU f;
  f.u[0] = *reinterpret_cast<const uint4*>(rowk0 + 2 * kb);
  f.u[1] = *reinterpret_cast<const uint4*>(rowk0 + 2 * kb + 8);
  return f.v;
}
// B-fragment: per-lane col, 16 contiguous K (caller adds 16*(lane>=16))
static __device__ __forceinline__ v16bf load_b_frag(const __bf16* p) {
  FragU f;
  f.u[0] = reinterpret_cast<const uint4*>(p)[0];
  f.u[1] = reinterpret_cast<const uint4*>(p)[1];
  return f.v;
}

// ---------------- prep kernels ----------------
__global__ __launch_bounds__(256) void cvt_f32_bf16(const float* __restrict__ s,
                                                    __bf16* __restrict__ d, int n) {
  int i = blockIdx.x * 256 + threadIdx.x;
  if (i < n) d[i] = (__bf16)s[i];
}

__global__ __launch_bounds__(256) void bn_affine(const float* __restrict__ sc,
                                                 const float* __restrict__ bi,
                                                 const float* __restrict__ mn,
                                                 const float* __restrict__ vr,
                                                 float* __restrict__ a,
                                                 float* __restrict__ c, int n) {
  int i = blockIdx.x * 256 + threadIdx.x;
  if (i < n) {
    float av = sc[i] * rsqrtf(vr[i] + 1e-5f);
    a[i] = av;
    c[i] = bi[i] - mn[i] * av;
  }
}

// in: [B][C][4096] f32  ->  out: [B][4096][C] bf16   (C multiple of 32)
__global__ __launch_bounds__(256) void transpose_to_bf16T(const float* __restrict__ in,
                                                          __bf16* __restrict__ out, int Cdim) {
  __shared__ float t[32][33];
  int b = blockIdx.z;
  int nb = blockIdx.x * 32, cb = blockIdx.y * 32;
  int tx = threadIdx.x, ty = threadIdx.y;       // (32,8)
  #pragma unroll
  for (int i = 0; i < 4; ++i) {
    int c = cb + ty + i * 8;
    t[ty + i * 8][tx] = in[((size_t)b * Cdim + c) * NPIX + nb + tx];
  }
  __syncthreads();
  #pragma unroll
  for (int i = 0; i < 4; ++i) {
    int n = nb + ty + i * 8;
    out[((size_t)b * NPIX + n) * Cdim + cb + tx] = (__bf16)t[tx][ty + i * 8];
  }
}

// 2x bilinear upsample (align_corners=False, half-pixel, edge clamp):
// low [B][512][32][32] f32 -> lowT [B][4096][512] bf16
__global__ __launch_bounds__(256) void upsample_lowT(const float* __restrict__ low,
                                                     __bf16* __restrict__ lowT) {
  int n = blockIdx.x & (NPIX - 1);
  int b = blockIdx.x >> 12;
  int y = n >> 6, x = n & 63;
  float sy = 0.5f * y - 0.25f;
  float sx = 0.5f * x - 0.25f;
  int yi = (int)floorf(sy), xi = (int)floorf(sx);
  float fy = sy - (float)yi, fx = sx - (float)xi;
  int y0 = max(yi, 0), y1 = min(yi + 1, 31);
  int x0 = max(xi, 0), x1 = min(xi + 1, 31);
  float w00 = (1.f - fy) * (1.f - fx), w01 = (1.f - fy) * fx;
  float w10 = fy * (1.f - fx),         w11 = fy * fx;
  const float* lb = low + (size_t)b * LDCH * 1024;
  __bf16* ob = lowT + ((size_t)b * NPIX + n) * LDCH;
  for (int c = threadIdx.x; c < LDCH; c += 256) {
    const float* p = lb + (size_t)c * 1024;
    float v = w00 * p[y0 * 32 + x0] + w01 * p[y0 * 32 + x1]
            + w10 * p[y1 * 32 + x0] + w11 * p[y1 * 32 + x1];
    ob[c] = (__bf16)v;
  }
}

// ---------------- generic NT GEMM: C[m][n] = sum_k A[m][k]*Bs[n][k] ----------------
// A: [M][K] bf16 (weights).  Bs: per batch [4096][K] bf16.
// EPI 0: y=a*x+c (opt relu), write transposed bf16 [4096][M]
// EPI 1: y=a*x+c (opt relu), write normal bf16 [M][4096]
// EPI 2: out = resid + gamma*relu(a*x+c), fp32 normal [M][4096]
enum { EPI_T = 0, EPI_N = 1, EPI_FINAL = 2 };

template <int EPI, bool RELU>
__global__ __launch_bounds__(256) void gemm_nt(const __bf16* __restrict__ A,
                                               const __bf16* __restrict__ Bs,
                                               int M, int K,
                                               const float* __restrict__ av,
                                               const float* __restrict__ cv,
                                               void* __restrict__ Cout,
                                               const float* __restrict__ resid,
                                               const float* __restrict__ gamma) {
  const int tid = threadIdx.x;
  const int w = tid >> 5, lane = tid & 31, lx = lane & 15, hh = lane >> 4;
  const int b  = blockIdx.z;
  const int m0 = blockIdx.x * 32 + (w & 1) * 16;
  const int n0 = blockIdx.y * 256 + (w >> 1) * 64;
  const int kb = hh * 8;
  v8f acc[4] = {};
  const __bf16* Arow = A + (size_t)(m0 + lx) * K;
  const __bf16* Bb   = Bs + (size_t)b * NPIX * K;
  for (int k0 = 0; k0 < K; k0 += 32) {
    v16bf af = load_a_frag(Arow + k0, kb);
    #pragma unroll
    for (int j = 0; j < 4; ++j) {
      int n = n0 + j * 16 + lx;
      v16bf bf_ = load_b_frag(Bb + (size_t)n * K + k0 + hh * 16);
      acc[j] = wmma_bf16(af, bf_, acc[j]);
    }
  }
  float ar[8], cr[8];
  #pragma unroll
  for (int r = 0; r < 8; ++r) {
    int row = m0 + hh * 8 + r;
    ar[r] = av ? av[row] : 1.0f;
    cr[r] = cv ? cv[row] : 0.0f;
  }
  if (EPI == EPI_T) {
    __bf16* out = (__bf16*)Cout;
    #pragma unroll
    for (int j = 0; j < 4; ++j) {
      int n = n0 + j * 16 + lx;
      Pack8 p;
      #pragma unroll
      for (int r = 0; r < 8; ++r) {
        float y = acc[j][r] * ar[r] + cr[r];
        if (RELU) y = fmaxf(y, 0.0f);
        p.e[r] = (__bf16)y;
      }
      *reinterpret_cast<uint4*>(out + ((size_t)b * NPIX + n) * M + m0 + hh * 8) = p.q;
    }
  } else if (EPI == EPI_N) {
    __bf16* out = (__bf16*)Cout;
    #pragma unroll
    for (int j = 0; j < 4; ++j) {
      int n = n0 + j * 16 + lx;
      #pragma unroll
      for (int r = 0; r < 8; ++r) {
        int row = m0 + hh * 8 + r;
        float y = acc[j][r] * ar[r] + cr[r];
        if (RELU) y = fmaxf(y, 0.0f);
        out[((size_t)b * M + row) * NPIX + n] = (__bf16)y;
      }
    }
  } else {
    float* out = (float*)Cout;
    float g = gamma[0];
    #pragma unroll
    for (int j = 0; j < 4; ++j) {
      int n = n0 + j * 16 + lx;
      #pragma unroll
      for (int r = 0; r < 8; ++r) {
        int row = m0 + hh * 8 + r;
        float y = acc[j][r] * ar[r] + cr[r];
        y = fmaxf(y, 0.0f);
        size_t idx = ((size_t)b * M + row) * NPIX + n;
        out[idx] = resid[idx] + g * y;
      }
    }
  }
}

// ---------------- softmax row stats (online max / sum-exp) ----------------
__global__ __launch_bounds__(256) void attn_stats(const __bf16* __restrict__ qT,
                                                  const __bf16* __restrict__ kT,
                                                  float* __restrict__ smax,
                                                  float* __restrict__ sinv) {
  const int tid = threadIdx.x;
  const int w = tid >> 5, lane = tid & 31, lx = lane & 15, hh = lane >> 4;
  const int b  = blockIdx.y;
  const int n0 = blockIdx.x * 128 + w * 16;
  const int kb = hh * 8;
  const __bf16* qb  = qT + (size_t)b * NPIX * EDIM;
  const __bf16* kbp = kT + (size_t)b * NPIX * EDIM;
  const __bf16* qrow = qb + (size_t)(n0 + lx) * EDIM;
  v16bf a0 = load_a_frag(qrow, kb);
  v16bf a1 = load_a_frag(qrow + 32, kb);
  float mrun[8], lrun[8];
  #pragma unroll
  for (int r = 0; r < 8; ++r) { mrun[r] = -1e30f; lrun[r] = 0.0f; }

  for (int m0 = 0; m0 < NPIX; m0 += 32) {
    v8f s[2];
    #pragma unroll
    for (int t = 0; t < 2; ++t) {
      int m = m0 + t * 16 + lx;
      const __bf16* kr = kbp + (size_t)m * EDIM + hh * 16;
      v8f a = {};
      a = wmma_bf16(a0, load_b_frag(kr), a);
      a = wmma_bf16(a1, load_b_frag(kr + 32), a);
      s[t] = a;
    }
    #pragma unroll
    for (int r = 0; r < 8; ++r) {
      float v = fmaxf(s[0][r], s[1][r]);
      #pragma unroll
      for (int off = 8; off >= 1; off >>= 1) v = fmaxf(v, __shfl_xor(v, off, 16));
      float mnew = fmaxf(mrun[r], v);
      float ssum = __expf(s[0][r] - mnew) + __expf(s[1][r] - mnew);
      #pragma unroll
      for (int off = 8; off >= 1; off >>= 1) ssum += __shfl_xor(ssum, off, 16);
      lrun[r] = lrun[r] * __expf(mrun[r] - mnew) + ssum;
      mrun[r] = mnew;
    }
  }
  if (lx == 0) {
    #pragma unroll
    for (int r = 0; r < 8; ++r) {
      int n = n0 + hh * 8 + r;
      smax[(size_t)b * NPIX + n] = mrun[r];
      sinv[(size_t)b * NPIX + n] = 1.0f / lrun[r];
    }
  }
}

// ---------------- P*V passes ----------------
// COLSTATS=false (out_high): rowT=qT, colT=kT, V=vh; stats indexed by A-rows (queries).
//   out[c][n] = sum_m P[n,m] vh[c,m]  -> C rows=c, cols=n, stored catT[n][c].
// COLSTATS=true  (out_low):  rowT=kT, colT=qT, V=vl; stats indexed by B-cols (queries).
//   out[c][m] = sum_n P[n,m] vl[c,n]  -> C rows=c, cols=m, stored catT[m][256+c].
template <bool COLSTATS>
__global__ __launch_bounds__(256) void attn_pv(const __bf16* __restrict__ rowT,
                                               const __bf16* __restrict__ colT,
                                               const __bf16* __restrict__ V,
                                               const float* __restrict__ smax,
                                               const float* __restrict__ sinv,
                                               __bf16* __restrict__ catT, int catoff) {
  __shared__ __attribute__((aligned(16))) __bf16 vs[ODIM * 32];   // V chunk [c][32]
  __shared__ __attribute__((aligned(16))) __bf16 ps[8][16 * 32];  // per-wave P tile
  const int tid = threadIdx.x;
  const int w = tid >> 5, lane = tid & 31, lx = lane & 15, hh = lane >> 4;
  const int b  = blockIdx.y;
  const int r0 = blockIdx.x * 128 + w * 16;      // this wave's 16 A-rows
  const int kb = hh * 8;
  const __bf16* rb  = rowT + (size_t)b * NPIX * EDIM;
  const __bf16* cbp = colT + (size_t)b * NPIX * EDIM;
  const __bf16* vb  = V + (size_t)b * ODIM * NPIX;
  const float* smb = smax + (size_t)b * NPIX;
  const float* sib = sinv + (size_t)b * NPIX;
  const __bf16* arow = rb + (size_t)(r0 + lx) * EDIM;
  v16bf a0 = load_a_frag(arow, kb);
  v16bf a1 = load_a_frag(arow + 32, kb);

  float rmax[8], rinv[8];
  if constexpr (!COLSTATS) {
    #pragma unroll
    for (int r = 0; r < 8; ++r) {
      int n = r0 + hh * 8 + r;
      rmax[r] = smb[n];
      rinv[r] = sib[n];
    }
  }
  v8f acc[16] = {};

  for (int c0 = 0; c0 < NPIX; c0 += 32) {
    __syncthreads();
    { // stage V chunk: vs[c][0..31] = V[c][c0..c0+31]  (coalesced)
      const __bf16* src = vb + (size_t)tid * NPIX + c0;
      uint4* d = reinterpret_cast<uint4*>(vs + tid * 32);
      d[0] = reinterpret_cast<const uint4*>(src)[0];
      d[1] = reinterpret_cast<const uint4*>(src)[1];
      __builtin_prefetch(src + 32, 0, 0);   // global_prefetch_b8 for next chunk
    }
    __syncthreads();

    // energy tiles: A-rows x 32 chunk cols
    #pragma unroll
    for (int t = 0; t < 2; ++t) {
      int cc = c0 + t * 16 + lx;
      const __bf16* kr = cbp + (size_t)cc * EDIM + hh * 16;
      v8f s = {};
      s = wmma_bf16(a0, load_b_frag(kr), s);
      s = wmma_bf16(a1, load_b_frag(kr + 32), s);
      float mx = 0.f, iv = 0.f;
      if constexpr (COLSTATS) { mx = smb[cc]; iv = sib[cc]; }
      #pragma unroll
      for (int r = 0; r < 8; ++r) {
        float p;
        if constexpr (COLSTATS) p = __expf(s[r] - mx) * iv;
        else                    p = __expf(s[r] - rmax[r]) * rinv[r];
        ps[w][(r + 8 * hh) * 32 + t * 16 + lx] = (__bf16)p;
      }
    }
    asm volatile("s_wait_dscnt 0x0" ::: "memory");  // wave-local LDS RAW fence

    // B-fragment from P: per-lane A-row (l&15), contiguous 16 chunk-K per half
    v16bf pf = load_b_frag(&ps[w][lx * 32 + hh * 16]);
    #pragma unroll
    for (int j = 0; j < 16; ++j) {
      v16bf va = load_a_frag(vs + (j * 16 + lx) * 32, kb);  // A = V rows c
      acc[j] = wmma_bf16(va, pf, acc[j]);
    }
  }

  // store: per lane fixed output column (r0+lx); rows c packed 8 at a time
  __bf16* dst = catT + ((size_t)b * NPIX + (r0 + lx)) * 512 + catoff;
  #pragma unroll
  for (int j = 0; j < 16; ++j) {
    Pack8 p;
    #pragma unroll
    for (int r = 0; r < 8; ++r) p.e[r] = (__bf16)acc[j][r];
    *reinterpret_cast<uint4*>(dst + j * 16 + hh * 8) = p.q;
  }
}

// ---------------- launcher ----------------
extern "C" void kernel_launch(void* const* d_in, const int* in_sizes, int n_in,
                              void* d_out, int out_size, void* d_ws, size_t ws_size,
                              hipStream_t stream) {
  (void)in_sizes; (void)n_in; (void)out_size; (void)ws_size;
  const float* high  = (const float*)d_in[0];
  const float* low   = (const float*)d_in[1];
  const float* W_high = (const float*)d_in[2];
  const float* bnh_s = (const float*)d_in[3];
  const float* bnh_b = (const float*)d_in[4];
  const float* bnh_m = (const float*)d_in[5];
  const float* bnh_v = (const float*)d_in[6];
  const float* W_low = (const float*)d_in[7];
  const float* bnl_s = (const float*)d_in[8];
  const float* bnl_b = (const float*)d_in[9];
  const float* bnl_m = (const float*)d_in[10];
  const float* bnl_v = (const float*)d_in[11];
  const float* W_q  = (const float*)d_in[12];
  const float* b_q  = (const float*)d_in[13];
  const float* W_k  = (const float*)d_in[14];
  const float* b_k  = (const float*)d_in[15];
  const float* W_vh = (const float*)d_in[16];
  const float* b_vh = (const float*)d_in[17];
  const float* W_vl = (const float*)d_in[18];
  const float* b_vl = (const float*)d_in[19];
  const float* W_out = (const float*)d_in[20];
  const float* bno_s = (const float*)d_in[21];
  const float* bno_b = (const float*)d_in[22];
  const float* bno_m = (const float*)d_in[23];
  const float* bno_v = (const float*)d_in[24];
  const float* gamma = (const float*)d_in[25];

  char* wsb = (char*)d_ws;
  size_t off = 0;
  auto alloc = [&](size_t bytes) -> void* {
    off = (off + 255) & ~(size_t)255;
    void* p = wsb + off;
    off += bytes;
    return p;
  };
  __bf16* highT = (__bf16*)alloc((size_t)BB * NPIX * HDCH * 2);
  __bf16* lowT  = (__bf16*)alloc((size_t)BB * NPIX * LDCH * 2);
  __bf16* heT   = (__bf16*)alloc((size_t)BB * NPIX * EDIM * 2);
  __bf16* leT   = (__bf16*)alloc((size_t)BB * NPIX * EDIM * 2);
  __bf16* qT    = (__bf16*)alloc((size_t)BB * NPIX * EDIM * 2);
  __bf16* kT    = (__bf16*)alloc((size_t)BB * NPIX * EDIM * 2);
  __bf16* vh    = (__bf16*)alloc((size_t)BB * ODIM * NPIX * 2);
  __bf16* vl    = (__bf16*)alloc((size_t)BB * ODIM * NPIX * 2);
  __bf16* catT  = (__bf16*)alloc((size_t)BB * NPIX * 512 * 2);
  float*  smax  = (float*)alloc((size_t)BB * NPIX * 4);
  float*  sinv  = (float*)alloc((size_t)BB * NPIX * 4);
  __bf16* Whb  = (__bf16*)alloc(64 * 256 * 2);
  __bf16* Wlb  = (__bf16*)alloc(64 * 512 * 2);
  __bf16* Wqb  = (__bf16*)alloc(64 * 64 * 2);
  __bf16* Wkb  = (__bf16*)alloc(64 * 64 * 2);
  __bf16* Wvhb = (__bf16*)alloc(256 * 256 * 2);
  __bf16* Wvlb = (__bf16*)alloc(256 * 512 * 2);
  __bf16* Wob  = (__bf16*)alloc(256 * 512 * 2);
  float* ah = (float*)alloc(64 * 4);
  float* ch = (float*)alloc(64 * 4);
  float* al = (float*)alloc(64 * 4);
  float* cl = (float*)alloc(64 * 4);
  float* ao = (float*)alloc(256 * 4);
  float* co = (float*)alloc(256 * 4);

  auto cvt = [&](const float* s, __bf16* d, int n) {
    cvt_f32_bf16<<<(n + 255) / 256, 256, 0, stream>>>(s, d, n);
  };
  cvt(W_high, Whb, 64 * 256);
  cvt(W_low,  Wlb, 64 * 512);
  cvt(W_q,    Wqb, 64 * 64);
  cvt(W_k,    Wkb, 64 * 64);
  cvt(W_vh,   Wvhb, 256 * 256);
  cvt(W_vl,   Wvlb, 256 * 512);
  cvt(W_out,  Wob, 256 * 512);
  bn_affine<<<1, 256, 0, stream>>>(bnh_s, bnh_b, bnh_m, bnh_v, ah, ch, 64);
  bn_affine<<<1, 256, 0, stream>>>(bnl_s, bnl_b, bnl_m, bnl_v, al, cl, 64);
  bn_affine<<<1, 256, 0, stream>>>(bno_s, bno_b, bno_m, bno_v, ao, co, 256);

  upsample_lowT<<<BB * NPIX, 256, 0, stream>>>(low, lowT);
  transpose_to_bf16T<<<dim3(NPIX / 32, HDCH / 32, BB), dim3(32, 8), 0, stream>>>(high, highT, HDCH);

  dim3 g64(2, 16, BB), g256(8, 16, BB);
  // he = relu(bn(W_high @ high))  -> heT[n][64]
  gemm_nt<EPI_T, true><<<g64, 256, 0, stream>>>(Whb, highT, 64, 256, ah, ch, heT, nullptr, nullptr);
  // le = relu(bn(W_low @ low_up)) -> leT[n][64]
  gemm_nt<EPI_T, true><<<g64, 256, 0, stream>>>(Wlb, lowT, 64, 512, al, cl, leT, nullptr, nullptr);
  // q = W_q@he + b_q -> qT ; k = W_k@le + b_k -> kT
  gemm_nt<EPI_T, false><<<g64, 256, 0, stream>>>(Wqb, heT, 64, 64, nullptr, b_q, qT, nullptr, nullptr);
  gemm_nt<EPI_T, false><<<g64, 256, 0, stream>>>(Wkb, leT, 64, 64, nullptr, b_k, kT, nullptr, nullptr);
  // vh = W_vh@high + b_vh [256][4096] ; vl = W_vl@low_up + b_vl
  gemm_nt<EPI_N, false><<<g256, 256, 0, stream>>>(Wvhb, highT, 256, 256, nullptr, b_vh, vh, nullptr, nullptr);
  gemm_nt<EPI_N, false><<<g256, 256, 0, stream>>>(Wvlb, lowT, 256, 512, nullptr, b_vl, vl, nullptr, nullptr);

  dim3 ga(NPIX / 128, BB);
  attn_stats<<<ga, 256, 0, stream>>>(qT, kT, smax, sinv);
  attn_pv<false><<<ga, 256, 0, stream>>>(qT, kT, vh, smax, sinv, catT, 0);    // out_high -> catT[:, :256]
  attn_pv<true ><<<ga, 256, 0, stream>>>(kT, qT, vl, smax, sinv, catT, 256);  // out_low  -> catT[:, 256:]

  // final: out = high + gamma * relu(bn(W_out @ cat))
  gemm_nt<EPI_FINAL, true><<<g256, 256, 0, stream>>>(Wob, catT, 256, 512, ao, co,
                                                     d_out, high, gamma);
}
